// MHA_75600014344476
// MI455X (gfx1250) — matrix operand order repaired and checked by reading, not compiled, not run on hardware
//
#include <hip/hip_runtime.h>

// ---------------- types ----------------
typedef __bf16 bf;
typedef __attribute__((ext_vector_type(16))) __bf16 v16bf;
typedef __attribute__((ext_vector_type(8)))  __bf16 v8bf;
typedef __attribute__((ext_vector_type(8)))  float  v8f;

#define DIM 2048
#define NH 16
#define HD 128
#define BATCH 2
#define SEQ 2048
#define MROWS (BATCH * SEQ)     // 4096
#define E3 (3 * DIM)            // 6144

__device__ __forceinline__ bf f2bf(float f) {
    unsigned u = __float_as_uint(f);
    u += 0x7fffu + ((u >> 16) & 1u);            // round-to-nearest-even
    unsigned short h = (unsigned short)(u >> 16);
    return __builtin_bit_cast(bf, h);
}
__device__ __forceinline__ float bf2f(bf b) {
    unsigned short h = __builtin_bit_cast(unsigned short, b);
    return __uint_as_float(((unsigned)h) << 16);
}

// Build an A-operand fragment (16-bit A 16x32 layout): lane holds row (lane&15),
// K chunks {half*16 .. half*16+7} and {half*16+8 .. half*16+15} (contiguous k order, same as B).
__device__ __forceinline__ v16bf ldA(const bf* lo, const bf* hi) {
    v8bf a = *(const v8bf*)lo;
    v8bf b = *(const v8bf*)hi;
    v16bf r;
#pragma unroll
    for (int j = 0; j < 8; ++j) { r[j] = a[j]; r[j + 8] = b[j]; }
    return r;
}

__device__ __forceinline__ v8f wmma_bf16(v16bf a, v16bf b, v8f c) {
    return __builtin_amdgcn_wmma_f32_16x16x32_bf16(
        /*neg_a=*/false, a, /*neg_b=*/false, b,
        /*c_mod=*/(short)0, c, /*reuse_a=*/false, /*reuse_b=*/false);
}

// ---------------- f32 -> bf16 convert ----------------
__global__ void cvt_f32_bf16(const float* __restrict__ in, bf* __restrict__ out, int n) {
    int i = blockIdx.x * blockDim.x + threadIdx.x;
    if (i < n) out[i] = f2bf(in[i]);
}

// ---------------- GEMM: C[M,N] = A[M,K](bf16,row) x W[N,K](bf16,row)^T -----------
// wave tile 32(M) x 64(N); block = 4 waves along N; grid = (M/32, N/256)
template <bool OUTF32>
__global__ __launch_bounds__(128) void gemm_bf16_kn(
    const bf* __restrict__ A, const bf* __restrict__ W,
    float* __restrict__ outF, bf* __restrict__ outB,
    int Ndim, int K)
{
    const int lane = threadIdx.x & 31;
    const int wid  = threadIdx.x >> 5;
    const int half = lane >> 4;
    const int l16  = lane & 15;
    const int m0 = blockIdx.x * 32;
    const int n0 = blockIdx.y * 256 + wid * 64;

    v8f c[2][4];
#pragma unroll
    for (int mi = 0; mi < 2; ++mi)
#pragma unroll
        for (int nt = 0; nt < 4; ++nt)
#pragma unroll
            for (int j = 0; j < 8; ++j) c[mi][nt][j] = 0.0f;

    const bf* Arow0 = A + (size_t)(m0 + l16) * K;
    const bf* Arow1 = A + (size_t)(m0 + 16 + l16) * K;
    const bf* Wr[4];
#pragma unroll
    for (int nt = 0; nt < 4; ++nt)
        Wr[nt] = W + (size_t)(n0 + nt * 16 + l16) * K + half * 16;

    for (int k0 = 0; k0 < K; k0 += 32) {
        if (k0 + 64 <= K) {                        // prefetch next K block
            __builtin_prefetch(Wr[0] + k0 + 32, 0, 1);
            __builtin_prefetch(Arow0 + k0 + 32, 0, 1);
        }
        v16bf a0 = ldA(Arow0 + k0 + half * 16, Arow0 + k0 + half * 16 + 8);
        v16bf a1 = ldA(Arow1 + k0 + half * 16, Arow1 + k0 + half * 16 + 8);
        v16bf b0 = *(const v16bf*)(Wr[0] + k0);
        v16bf b1 = *(const v16bf*)(Wr[1] + k0);
        v16bf b2 = *(const v16bf*)(Wr[2] + k0);
        v16bf b3 = *(const v16bf*)(Wr[3] + k0);
        c[0][0] = wmma_bf16(a0, b0, c[0][0]);
        c[0][1] = wmma_bf16(a0, b1, c[0][1]);
        c[0][2] = wmma_bf16(a0, b2, c[0][2]);
        c[0][3] = wmma_bf16(a0, b3, c[0][3]);
        c[1][0] = wmma_bf16(a1, b0, c[1][0]);
        c[1][1] = wmma_bf16(a1, b1, c[1][1]);
        c[1][2] = wmma_bf16(a1, b2, c[1][2]);
        c[1][3] = wmma_bf16(a1, b3, c[1][3]);
    }

#pragma unroll
    for (int mi = 0; mi < 2; ++mi)
#pragma unroll
        for (int nt = 0; nt < 4; ++nt)
#pragma unroll
            for (int i = 0; i < 8; ++i) {
                int row = m0 + mi * 16 + half * 8 + i;   // C layout: VGPR i -> M = half*8+i
                int col = n0 + nt * 16 + l16;            // lane -> N = lane&15
                float v = c[mi][nt][i];
                if (OUTF32) outF[(size_t)row * Ndim + col] = v;
                else        outB[(size_t)row * Ndim + col] = f2bf(v);
            }
}

// ---------------- RoPE + layout scatter ----------------
// qkv bf16 [4096, 6144] -> qb/kb [b,h,T,128] (rotary applied), vT [b,h,128,T]
__global__ void rope_scatter(const bf* __restrict__ qkv, const float* __restrict__ fc,
                             bf* __restrict__ qb, bf* __restrict__ kb, bf* __restrict__ vT)
{
    int idx = blockIdx.x * blockDim.x + threadIdx.x;   // [0, 2*16*2048*64)
    int i = idx & 63;
    int t = (idx >> 6) & 2047;
    int h = (idx >> 17) & 15;
    int b = idx >> 21;
    if (b >= BATCH) return;
    int d0 = 2 * i;
    size_t row = (size_t)(b * SEQ + t) * E3;
    int col = h * HD + d0;
    float q0 = bf2f(qkv[row + col]),            q1 = bf2f(qkv[row + col + 1]);
    float k0 = bf2f(qkv[row + DIM + col]),      k1 = bf2f(qkv[row + DIM + col + 1]);
    float v0 = bf2f(qkv[row + 2 * DIM + col]),  v1 = bf2f(qkv[row + 2 * DIM + col + 1]);
    float f = fc[t * HD + d0];                  // freqs repeated pairwise
    float s, cth;
    __sincosf(f, &s, &cth);
    size_t bh = (size_t)(b * NH + h);
    size_t qo = (bh * SEQ + t) * HD + d0;
    qb[qo]     = f2bf(q0 * cth - q1 * s);
    qb[qo + 1] = f2bf(q1 * cth + q0 * s);
    kb[qo]     = f2bf(k0 * cth - k1 * s);
    kb[qo + 1] = f2bf(k1 * cth + k0 * s);
    vT[(bh * HD + d0) * SEQ + t]     = f2bf(v0);
    vT[(bh * HD + d0 + 1) * SEQ + t] = f2bf(v1);
}

// ---------------- causal flash attention ----------------
// one 16-query tile per wave; 4 waves/block; online softmax; P staged via LDS
__global__ __launch_bounds__(128) void attn_kernel(
    const bf* __restrict__ qb, const bf* __restrict__ kb, const bf* __restrict__ vT,
    bf* __restrict__ ob)
{
    __shared__ __bf16 lds[4][16][40];   // stride 40 elems (80B): conflict-free b128 reload
    const int lane = threadIdx.x & 31;
    const int wid  = threadIdx.x >> 5;
    const int half = lane >> 4;
    const int l16  = lane & 15;
    int gw = blockIdx.x * 4 + wid;      // 0..4095
    int qt = gw & 127;
    int h  = (gw >> 7) & 15;
    int b  = gw >> 11;
    int m0 = qt * 16;
    size_t bh = (size_t)(b * NH + h);
    const bf* qbase = qb + bh * (size_t)SEQ * HD;
    const bf* kbase = kb + bh * (size_t)SEQ * HD;
    const bf* vbase = vT + bh * (size_t)HD * SEQ;

    v8f o[8];
#pragma unroll
    for (int nt = 0; nt < 8; ++nt)
#pragma unroll
        for (int j = 0; j < 8; ++j) o[nt][j] = 0.0f;
    float mrow[8], lrow[8];
#pragma unroll
    for (int i = 0; i < 8; ++i) { mrow[i] = -3.0e38f; lrow[i] = 0.0f; }

    const float scale = 0.08838834764831845f;   // 1/sqrt(128)

    v16bf qf[4];                                 // whole q tile held in registers
#pragma unroll
    for (int dk4 = 0; dk4 < 4; ++dk4) {
        const bf* r = qbase + (size_t)(m0 + l16) * HD + dk4 * 32;
        qf[dk4] = ldA(r + half * 16, r + half * 16 + 8);
    }

    const int kend = m0 + 16;                    // causal bound
    for (int kc = 0; kc < kend; kc += 32) {
        v8f s0, s1;
#pragma unroll
        for (int j = 0; j < 8; ++j) { s0[j] = 0.0f; s1[j] = 0.0f; }
#pragma unroll
        for (int dk4 = 0; dk4 < 4; ++dk4) {
            int dk = dk4 * 32;
            v16bf kf0 = *(const v16bf*)(kbase + (size_t)(kc + l16) * HD + dk + half * 16);
            v16bf kf1 = *(const v16bf*)(kbase + (size_t)(kc + 16 + l16) * HD + dk + half * 16);
            s0 = wmma_bf16(qf[dk4], kf0, s0);
            s1 = wmma_bf16(qf[dk4], kf1, s1);
        }
        int key0 = kc + l16, key1 = kc + 16 + l16;
#pragma unroll
        for (int i = 0; i < 8; ++i) {
            int qrow = m0 + half * 8 + i;
            float x0 = s0[i] * scale; if (key0 > qrow) x0 = -3.0e38f;
            float x1 = s1[i] * scale; if (key1 > qrow) x1 = -3.0e38f;
            float rm = fmaxf(x0, x1);            // row-max across 16 lanes of this half
            rm = fmaxf(rm, __shfl_xor(rm, 1, 32));
            rm = fmaxf(rm, __shfl_xor(rm, 2, 32));
            rm = fmaxf(rm, __shfl_xor(rm, 4, 32));
            rm = fmaxf(rm, __shfl_xor(rm, 8, 32));
            float mnew = fmaxf(mrow[i], rm);
            float p0 = __expf(x0 - mnew);
            float p1 = __expf(x1 - mnew);
            float rs = p0 + p1;
            rs += __shfl_xor(rs, 1, 32);
            rs += __shfl_xor(rs, 2, 32);
            rs += __shfl_xor(rs, 4, 32);
            rs += __shfl_xor(rs, 8, 32);
            float corr = __expf(mrow[i] - mnew);
            lrow[i] = lrow[i] * corr + rs;
            mrow[i] = mnew;
#pragma unroll
            for (int nt = 0; nt < 8; ++nt) o[nt][i] *= corr;
            lds[wid][half * 8 + i][l16]      = f2bf(p0);   // C-layout store
            lds[wid][half * 8 + i][16 + l16] = f2bf(p1);
        }
        asm volatile("s_wait_dscnt 0" ::: "memory");        // wave-local LDS fence
        v16bf pf = ldA(&lds[wid][l16][half * 16], &lds[wid][l16][half * 16 + 8]);
#pragma unroll
        for (int nt = 0; nt < 8; ++nt) {
            v16bf vf = *(const v16bf*)(vbase + (size_t)(nt * 16 + l16) * SEQ + kc + half * 16);
            o[nt] = wmma_bf16(pf, vf, o[nt]);
        }
    }

#pragma unroll
    for (int i = 0; i < 8; ++i) {
        float inv = 1.0f / lrow[i];
        int t = m0 + half * 8 + i;
        size_t base = (((size_t)b * SEQ + t) * NH + h) * HD;   // ob: [B,T,h,hd]
#pragma unroll
        for (int nt = 0; nt < 8; ++nt)
            ob[base + nt * 16 + l16] = f2bf(o[nt][i] * inv);
    }
}

// ---------------- launcher ----------------
extern "C" void kernel_launch(void* const* d_in, const int* in_sizes, int n_in,
                              void* d_out, int out_size, void* d_ws, size_t ws_size,
                              hipStream_t stream) {
    const float* x    = (const float*)d_in[0];   // [2,2048,2048]
    const float* fc   = (const float*)d_in[1];   // [2048,128]
    const float* Wqkv = (const float*)d_in[2];   // [6144,2048]
    const float* Wout = (const float*)d_in[3];   // [2048,2048]
    float* out = (float*)d_out;

    unsigned char* ws = (unsigned char*)d_ws;
    bf* xb    = (bf*)(ws + 0);                                // 16 MB
    bf* wqkvb = (bf*)(ws + (size_t)16 * 1024 * 1024);         // 24 MB
    bf* woutb = (bf*)(ws + (size_t)40 * 1024 * 1024);         //  8 MB
    bf* qkvb  = (bf*)(ws + (size_t)48 * 1024 * 1024);         // 48 MB
    bf* qbuf  = (bf*)(ws + (size_t)96 * 1024 * 1024);         // 16 MB
    bf* kbuf  = (bf*)(ws + (size_t)112 * 1024 * 1024);        // 16 MB
    bf* vT    = (bf*)(ws + (size_t)128 * 1024 * 1024);        // 16 MB
    bf* ob    = (bf*)(ws + (size_t)144 * 1024 * 1024);        // 16 MB  (total 160 MB)

    // 1) convert inputs to bf16
    {
        int n = MROWS * DIM;                  // 8,388,608
        cvt_f32_bf16<<<(n + 255) / 256, 256, 0, stream>>>(x, xb, n);
        n = E3 * DIM;                         // 12,582,912
        cvt_f32_bf16<<<(n + 255) / 256, 256, 0, stream>>>(Wqkv, wqkvb, n);
        n = DIM * DIM;                        // 4,194,304
        cvt_f32_bf16<<<(n + 255) / 256, 256, 0, stream>>>(Wout, woutb, n);
    }
    // 2) QKV projection: [4096,2048] x [6144,2048]^T -> bf16 [4096,6144]
    gemm_bf16_kn<false><<<dim3(MROWS / 32, E3 / 256), 128, 0, stream>>>(
        xb, wqkvb, nullptr, qkvb, E3, DIM);
    // 3) rotary + scatter to attention layouts
    rope_scatter<<<(BATCH * NH * SEQ * 64) / 256, 256, 0, stream>>>(
        qkvb, fc, qbuf, kbuf, vT);
    // 4) causal flash attention (4096 query tiles, 4 waves/block)
    attn_kernel<<<(BATCH * NH * (SEQ / 16)) / 4, 128, 0, stream>>>(qbuf, kbuf, vT, ob);
    // 5) output projection: [4096,2048] x [2048,2048]^T -> f32 d_out
    gemm_bf16_kn<true><<<dim3(MROWS / 32, DIM / 256), 128, 0, stream>>>(
        ob, woutb, out, nullptr, DIM, DIM);
}
